// VQVAE_10608569221657
// MI455X (gfx1250) — matrix operand order repaired and checked by reading, not compiled, not run on hardware
//
#include <hip/hip_runtime.h>
#include <hip/hip_bf16.h>
#include <math.h>

typedef unsigned short u16;
typedef unsigned int   u32;

typedef __attribute__((ext_vector_type(16))) __bf16 v16bf;
typedef __attribute__((ext_vector_type(8)))  float  v8f;

union AFrag { uint4 q[2]; u16 u[16]; v16bf v; };
union BFrag { u32 d[8]; u16 u[16]; v16bf v; };

__device__ __forceinline__ u16 f2bf(float f) {
  u32 u = __float_as_uint(f);
  u32 r = u + 0x7FFFu + ((u >> 16) & 1u);   // round-to-nearest-even
  return (u16)(r >> 16);
}

// Stage one 32x32 bf16 weight chunk (rows m0..m0+31, cols k0..k0+31) into LDS
// with the CDNA5 async copy engine (4 x global_load_async_to_lds_b128, 16B/lane).
// LDS layout matches global chunk: [r][c] at ldsBase + (r*32+c)*2.
__device__ __forceinline__ void stage_a_async(
    const u16* __restrict__ A, int rowBase /* = m0*Kpad */, int Kpad,
    int k0, u32 ldsBase, int lane)
{
#pragma unroll
  for (int q = 0; q < 4; ++q) {
    int linear = q * 256 + lane * 8;       // element index within 32x32 chunk
    int r = linear >> 5;
    int c = linear & 31;
    const u16* src = A + (rowBase + r * Kpad + k0 + c);
    u32 dst = ldsBase + (u32)((r * 32 + c) * 2);
    asm volatile("global_load_async_to_lds_b128 %0, %1, off"
                 :: "v"(dst), "v"(src) : "memory");
  }
}

// ---------------------------------------------------------------------------
// Implicit-GEMM conv, bf16 WMMA, f32 out. One wave/block, 16(M)x64(N) tile.
// Weights double-buffered through LDS via async copies (ASYNCcnt);
// activations gathered straight to VGPRs (LOADcnt). All hot-loop offsets are
// 32-bit so loads use the SGPR-base + 32-bit-voffset (GVS) form.
// KS4=1: 4x4 conv over halo-padded [Cin][Bn][Hin+2][Win+2] bf16 (Kpad==Cin*16).
// KS4=0: 1x1 GEMM over plain [K][N] bf16.
// ---------------------------------------------------------------------------
template <int KS4>
__global__ __launch_bounds__(32) void conv_gemm_k(
    const u16* __restrict__ A, const u16* __restrict__ Bact,
    const float* __restrict__ bias, float* __restrict__ Out,
    int Cout, int Nout, int Kpad,
    int Bn, int Hin, int Win, int Hout, int Wout,
    int stride, int pad)
{
  const int lane = threadIdx.x & 31;
  const int half = lane >> 4;
  const int ln   = lane & 15;
  const int m0   = blockIdx.y << 4;
  const int n0   = blockIdx.x << 6;

  const int Hp = Hin + 2, Wp = Win + 2;
  const int chStride = Bn * Hp * Wp;      // halo channel stride (elements)
  const int NinPlain = Bn * Hin * Win;    // plain channel stride (elements)

  __align__(16) __shared__ u16 As[2][512];   // 2 x (16x32) bf16 double buffer
  const u32 lds0 = (u32)(size_t)&As[0][0];
  const u32 lds1 = (u32)(size_t)&As[1][0];

  int nIdx[4];
  int off[4];
#pragma unroll
  for (int t = 0; t < 4; ++t) {
    int n = n0 + (t << 4) + ln;
    nIdx[t] = n;
    if (KS4) {
      const int HWout = Hout * Wout;
      int b  = n / HWout;
      int r  = n - b * HWout;
      int oy = r / Wout;
      int ox = r - oy * Wout;
      int pys = oy * stride - pad + 1;
      int pxs = ox * stride - pad + 1;     // even for s2p1 -> dword aligned
      off[t] = (b * Hp + pys) * Wp + pxs;
    } else {
      off[t] = 0;
    }
  }

  v8f acc[4];
#pragma unroll
  for (int t = 0; t < 4; ++t) acc[t] = (v8f){0.f,0.f,0.f,0.f,0.f,0.f,0.f,0.f};

  const int rowBase = m0 * Kpad;
  stage_a_async(A, rowBase, Kpad, 0, lds0, lane);   // prologue: chunk 0
  int buf = 0;

  for (int k0 = 0; k0 < Kpad; k0 += 32) {
    if (k0 + 32 < Kpad) {
      stage_a_async(A, rowBase, Kpad, k0 + 32, buf ? lds0 : lds1, lane);
      asm volatile("s_wait_asynccnt 0x4" ::: "memory");  // current chunk done
    } else {
      asm volatile("s_wait_asynccnt 0x0" ::: "memory");
    }

    // A fragment from LDS: lane(m=ln,half) holds K in [16h,16h+16)
    AFrag af;
    const u16* asrc = &As[buf][ln * 32 + (half << 4)];
    af.q[0] = *(const uint4*)asrc;
    af.q[1] = *(const uint4*)(asrc + 8);

    const int kb = k0 + (half << 4);   // this lane's 16 consecutive K values
#pragma unroll
    for (int t = 0; t < 4; ++t) {
      BFrag bfm;
      if (KS4) {
        const int ci = kb >> 4;        // kb multiple of 16 -> one channel
        const int eb = ci * chStride + off[t];
#pragma unroll
        for (int ky = 0; ky < 4; ++ky) {
          const int er = eb + ky * Wp;
          bfm.d[2 * ky]     = *(const u32*)(Bact + er);
          bfm.d[2 * ky + 1] = *(const u32*)(Bact + er + 2);
        }
      } else {
#pragma unroll
        for (int j = 0; j < 16; ++j)
          bfm.u[j] = Bact[(kb + j) * NinPlain + nIdx[t]];
      }
      acc[t] = __builtin_amdgcn_wmma_f32_16x16x32_bf16(
          false, af.v, false, bfm.v, (short)0, acc[t], false, false);
    }
    buf ^= 1;
  }

#pragma unroll
  for (int t = 0; t < 4; ++t) {
#pragma unroll
    for (int r = 0; r < 8; ++r) {
      int m = m0 + r + (half << 3);
      if (m < Cout) {
        float bs = bias ? bias[m] : 0.f;
        Out[m * Nout + n0 + (t << 4) + ln] = acc[t][r] + bs;
      }
    }
  }
}

// ---------------------------------------------------------------------------
// ConvTranspose2d(k=4,s=2,p=1) as 4 parity-class 2x2 convs over halo input.
// Apack: [4][Mpad][Cin*4] bf16, taps ascending in (iy,ix). Async-LDS weights.
// ---------------------------------------------------------------------------
__global__ __launch_bounds__(32) void deconv_gemm_k(
    const u16* __restrict__ Apack, const u16* __restrict__ Bact,
    const float* __restrict__ bias, float* __restrict__ Out,
    int Cout, int Mpad, int Cin,
    int Bn, int Hin, int Win, int Hout, int Wout)
{
  const int lane = threadIdx.x & 31;
  const int half = lane >> 4;
  const int ln   = lane & 15;
  const int p  = blockIdx.z;
  const int py = p >> 1, px = p & 1;
  const int Kp = Cin << 2;
  const u16* A = Apack + p * Mpad * Kp;
  const int m0 = blockIdx.y << 4;
  const int n0 = blockIdx.x << 6;
  const int H2 = Hout >> 1, W2 = Wout >> 1;
  const int HW2 = H2 * W2;
  const int Hp = Hin + 2, Wp = Win + 2;
  const int chStride = Bn * Hp * Wp;
  const int NoutT = Bn * Hout * Wout;

  __align__(16) __shared__ u16 As[2][512];
  const u32 lds0 = (u32)(size_t)&As[0][0];
  const u32 lds1 = (u32)(size_t)&As[1][0];

  int bb[4], y2a[4], x2a[4];
  int off[4];
#pragma unroll
  for (int t = 0; t < 4; ++t) {
    int n = n0 + (t << 4) + ln;
    int b  = n / HW2;
    int r  = n - b * HW2;
    int y2 = r / W2;
    int x2 = r - y2 * W2;
    bb[t] = b; y2a[t] = y2; x2a[t] = x2;
    off[t] = (b * Hp + (y2 + py)) * Wp + (x2 + px);
  }

  v8f acc[4];
#pragma unroll
  for (int t = 0; t < 4; ++t) acc[t] = (v8f){0.f,0.f,0.f,0.f,0.f,0.f,0.f,0.f};

  const int rowBase = m0 * Kp;
  stage_a_async(A, rowBase, Kp, 0, lds0, lane);
  int buf = 0;

  for (int k0 = 0; k0 < Kp; k0 += 32) {
    if (k0 + 32 < Kp) {
      stage_a_async(A, rowBase, Kp, k0 + 32, buf ? lds0 : lds1, lane);
      asm volatile("s_wait_asynccnt 0x4" ::: "memory");
    } else {
      asm volatile("s_wait_asynccnt 0x0" ::: "memory");
    }

    AFrag af;
    const u16* asrc = &As[buf][ln * 32 + (half << 4)];
    af.q[0] = *(const uint4*)asrc;
    af.q[1] = *(const uint4*)(asrc + 8);

    const int kb = k0 + (half << 4);   // multiple of 16 -> 4 channels x 4 taps
#pragma unroll
    for (int t = 0; t < 4; ++t) {
      BFrag bfm;
      const int ci0 = kb >> 2;
#pragma unroll
      for (int g = 0; g < 4; ++g) {
        const int eb = (ci0 + g) * chStride + off[t];
        bfm.u[4 * g + 0] = Bact[eb];
        bfm.u[4 * g + 1] = Bact[eb + 1];
        bfm.u[4 * g + 2] = Bact[eb + Wp];
        bfm.u[4 * g + 3] = Bact[eb + Wp + 1];
      }
      acc[t] = __builtin_amdgcn_wmma_f32_16x16x32_bf16(
          false, af.v, false, bfm.v, (short)0, acc[t], false, false);
    }
    buf ^= 1;
  }

#pragma unroll
  for (int t = 0; t < 4; ++t) {
    int oy = (y2a[t] << 1) + py;
    int ox = (x2a[t] << 1) + px;
    int nOut = (bb[t] * Hout + oy) * Wout + ox;
#pragma unroll
    for (int r = 0; r < 8; ++r) {
      int m = m0 + r + (half << 3);
      if (m < Cout) {
        float bs = bias ? bias[m] : 0.f;
        Out[m * NoutT + nOut] = acc[t][r] + bs;
      }
    }
  }
}

// ---------------------------------------------------------------------------
// Weight packing (f32 -> bf16, zero padded)
// ---------------------------------------------------------------------------
__global__ void pack_conv_k(const float* __restrict__ W, u16* __restrict__ Wp,
                            int Cout, int Ktot, int Kpad, int Mpad)
{
  long long i = (long long)blockIdx.x * 256 + threadIdx.x;
  long long total = (long long)Mpad * Kpad;
  if (i >= total) return;
  int co = (int)(i / Kpad);
  int kp = (int)(i - (long long)co * Kpad);
  float v = (co < Cout && kp < Ktot) ? W[(long long)co * Ktot + kp] : 0.f;
  Wp[i] = f2bf(v);
}

__global__ void pack_deconv_k(const float* __restrict__ W, u16* __restrict__ Wp,
                              int Cout, int Mpad, int Cin)
{
  int Kp = Cin << 2;
  long long i = (long long)blockIdx.x * 256 + threadIdx.x;
  long long per = (long long)Mpad * Kp;
  long long total = 4 * per;
  if (i >= total) return;
  int p  = (int)(i / per);
  long long r = i - (long long)p * per;
  int co = (int)(r / Kp);
  int k  = (int)(r - (long long)co * Kp);
  int ci  = k >> 2;
  int kyi = (k >> 1) & 1;
  int kxi = k & 1;
  int py = p >> 1, px = p & 1;
  int ky = (py == 0) ? (kyi ? 1 : 3) : (kyi ? 0 : 2);  // ascending iy order
  int kx = (px == 0) ? (kxi ? 1 : 3) : (kxi ? 0 : 2);  // ascending ix order
  float v = (co < Cout) ? W[(((long long)ci * Cout + co) * 4 + ky) * 4 + kx] : 0.f;
  Wp[i] = f2bf(v);
}

// ---------------------------------------------------------------------------
// Layout / conversion kernels
// ---------------------------------------------------------------------------
__global__ void zero_u16_k(u16* __restrict__ p, long long n)
{
  long long i = (long long)blockIdx.x * 256 + threadIdx.x;
  if (i < n) p[i] = 0;
}

__global__ void halo_zero_k(u16* __restrict__ act, int C, int Bn, int H, int W)
{
  int Hp = H + 2, Wp = W + 2;
  int per = 2 * Wp + 2 * H;
  long long i = (long long)blockIdx.x * 256 + threadIdx.x;
  long long total = (long long)C * Bn * per;
  if (i >= total) return;
  int e = (int)(i % per);
  long long cb = i / per;
  long long basep = cb * (long long)Hp * Wp;
  int y, xx;
  if (e < Wp)            { y = 0;      xx = e; }
  else if (e < 2 * Wp)   { y = Hp - 1; xx = e - Wp; }
  else { int rr = e - 2 * Wp; y = 1 + (rr >> 1); xx = (rr & 1) ? (Wp - 1) : 0; }
  act[basep + (long long)y * Wp + xx] = 0;
}

__global__ void nchw2halo_k(const float* __restrict__ X, u16* __restrict__ Y,
                            int Bn, int C, int H, int W)
{
  long long i = (long long)blockIdx.x * 256 + threadIdx.x;
  long long total = (long long)Bn * C * H * W;
  if (i >= total) return;
  int HW = H * W;
  int hw = (int)(i % HW);
  long long t = i / HW;
  int c = (int)(t % C);
  int b = (int)(t / C);
  int y = hw / W, xx = hw - y * W;
  int Hp = H + 2, Wp = W + 2;
  Y[(((long long)c * Bn + b) * Hp + y + 1) * Wp + xx + 1] = f2bf(X[i]);
}

__global__ void f32_to_bf16_k(const float* __restrict__ X, u16* __restrict__ Y, long long n)
{
  long long i = (long long)blockIdx.x * 256 + threadIdx.x;
  if (i < n) Y[i] = f2bf(X[i]);
}

__global__ void f32cn_to_halo_k(const float* __restrict__ X, u16* __restrict__ Y,
                                int Bn, int H, int W)
{
  int c = blockIdx.y;
  int n = blockIdx.x * 256 + threadIdx.x;
  int HW = H * W;
  int b = n / HW;
  int hw = n - b * HW;
  int y = hw / W, xx = hw - y * W;
  int Hp = H + 2, Wp = W + 2;
  long long N = (long long)Bn * HW;
  Y[(((long long)c * Bn + b) * Hp + y + 1) * Wp + xx + 1] = f2bf(X[(long long)c * N + n]);
}

// ---------------------------------------------------------------------------
// BatchNorm (training mode, biased var)
// ---------------------------------------------------------------------------
__global__ void bn_stats_k(const float* __restrict__ X, float* __restrict__ stats, int N)
{
  int c = blockIdx.x;
  const float* row = X + (long long)c * N;
  float s = 0.f, s2 = 0.f;
  for (int i = threadIdx.x; i < N; i += 256) {
    float v = row[i];
    s += v; s2 += v * v;
  }
  __shared__ float sh0[256], sh1[256];
  sh0[threadIdx.x] = s; sh1[threadIdx.x] = s2;
  __syncthreads();
  for (int st = 128; st; st >>= 1) {
    if (threadIdx.x < st) { sh0[threadIdx.x] += sh0[threadIdx.x + st];
                            sh1[threadIdx.x] += sh1[threadIdx.x + st]; }
    __syncthreads();
  }
  if (threadIdx.x == 0) {
    float m = sh0[0] / (float)N;
    float v = sh1[0] / (float)N - m * m;
    stats[2 * c]     = m;
    stats[2 * c + 1] = rsqrtf(v + 1e-5f);
  }
}

template <int HALO>
__global__ void bn_relu_k(const float* __restrict__ X, const float* __restrict__ stats,
                          const float* __restrict__ g, const float* __restrict__ be,
                          u16* __restrict__ Yb, int Bn, int H, int W)
{
  int c = blockIdx.y;
  int n = blockIdx.x * 256 + threadIdx.x;
  int HW = H * W;
  long long N = (long long)Bn * HW;
  float m = stats[2 * c], rs = stats[2 * c + 1];
  float y = g[c] * (X[(long long)c * N + n] - m) * rs + be[c];
  u16 v = f2bf(fmaxf(y, 0.f));
  if (HALO) {
    int b = n / HW;
    int hw = n - b * HW;
    int yy = hw / W, xx = hw - yy * W;
    int Hp = H + 2, Wp = W + 2;
    Yb[(((long long)c * Bn + b) * Hp + yy + 1) * Wp + xx + 1] = v;
  } else {
    Yb[(long long)c * N + n] = v;
  }
}

__global__ void bn_sig_out_k(const float* __restrict__ X, const float* __restrict__ stats,
                             const float* __restrict__ g, const float* __restrict__ be,
                             float* __restrict__ out, int N, int HW, int C)
{
  int c = blockIdx.y;
  int n = blockIdx.x * 256 + threadIdx.x;
  float m = stats[2 * c], rs = stats[2 * c + 1];
  float y = g[c] * (X[(long long)c * N + n] - m) * rs + be[c];
  y = 1.f / (1.f + expf(-y));
  int b  = n / HW;
  int hw = n - b * HW;
  out[((long long)b * C + c) * HW + hw] = y;
}

// ---------------------------------------------------------------------------
// Vector quantization
// ---------------------------------------------------------------------------
__global__ void cc_k(const float* __restrict__ cb, float* __restrict__ cc)
{
  int k = blockIdx.x * 256 + threadIdx.x;
  if (k >= 512) return;
  float s = 0.f;
  for (int c = 0; c < 128; ++c) { float v = cb[k * 128 + c]; s += v * v; }
  cc[k] = s;
}

__global__ void vq_argmin_k(const float* __restrict__ scores, const float* __restrict__ cc,
                            int* __restrict__ idx)
{
  int n = blockIdx.x * 256 + threadIdx.x;   // 2048 total
  float best = 3.4e38f;
  int bi = 0;
  for (int k = 0; k < 512; ++k) {
    float d = cc[k] - 2.f * scores[k * 2048 + n];
    if (d < best) { best = d; bi = k; }
  }
  idx[n] = bi;
}

__global__ void vq_gather_k(const float* __restrict__ cb, const int* __restrict__ idx,
                            u16* __restrict__ act)
{
  long long i = (long long)blockIdx.x * 256 + threadIdx.x;
  if (i >= 128LL * 2048) return;
  int c = (int)(i >> 11);
  int n = (int)(i & 2047);
  act[i] = f2bf(cb[idx[n] * 128 + c]);
}

__global__ void zero2_k(float* a) { a[0] = 0.f; a[1] = 0.f; }

__global__ void vq_loss_k(const float* __restrict__ qf, const float* __restrict__ cb,
                          const int* __restrict__ idx, float* __restrict__ acc)
{
  int n = blockIdx.x;      // 2048
  int c = threadIdx.x;     // 128
  float q  = qf[c * 2048 + n];
  float qt = cb[idx[n] * 128 + c];
  float d = qt - q;
  __shared__ float s1[128], s2[128];
  s1[c] = d * d;
  s2[c] = qt - q * q;
  __syncthreads();
  for (int st = 64; st; st >>= 1) {
    if (c < st) { s1[c] += s1[c + st]; s2[c] += s2[c + st]; }
    __syncthreads();
  }
  if (c == 0) { atomicAdd(&acc[0], s1[0]); atomicAdd(&acc[1], s2[0]); }
}

__global__ void final_loss_k(const float* __restrict__ acc, float* __restrict__ out)
{
  const float total = 262144.f;   // 2048 * 128
  out[0] = acc[1] / total + 0.25f * (acc[0] / total);
}

// ---------------------------------------------------------------------------
extern "C" void kernel_launch(void* const* d_in, const int* in_sizes, int n_in,
                              void* d_out, int out_size, void* d_ws, size_t ws_size,
                              hipStream_t stream)
{
  (void)in_sizes; (void)n_in; (void)out_size; (void)ws_size;
  const int Bn = 32;

  const float* cb_f = (const float*)d_in[0];
  const float* dec_b[5]; const float* dec_be[5]; const float* dec_g[5]; const float* dec_w[5];
  const float* enc_b[5]; const float* enc_be[5]; const float* enc_g[5]; const float* enc_w[5];
  for (int i = 0; i < 5; ++i) {
    dec_b[i]  = (const float*)d_in[1 + 4 * i];
    dec_be[i] = (const float*)d_in[2 + 4 * i];
    dec_g[i]  = (const float*)d_in[3 + 4 * i];
    dec_w[i]  = (const float*)d_in[4 + 4 * i];
    enc_b[i]  = (const float*)d_in[21 + 4 * i];
    enc_be[i] = (const float*)d_in[22 + 4 * i];
    enc_g[i]  = (const float*)d_in[23 + 4 * i];
    enc_w[i]  = (const float*)d_in[24 + 4 * i];
  }
  const float* post_b = (const float*)d_in[41];
  const float* post_w = (const float*)d_in[42];
  const float* pre_b  = (const float*)d_in[43];
  const float* pre_w  = (const float*)d_in[44];
  const float* x      = (const float*)d_in[45];

  static const int eCinP[5] = {4, 64, 128, 256, 512};   // layer-1 channel-padded
  static const int eCout[5] = {64, 128, 256, 512, 1024};
  static const int eHin[5]  = {256, 128, 64, 32, 16};
  static const int dCin[5]  = {1024, 512, 256, 128, 64};
  static const int dCout[5] = {512, 256, 128, 64, 3};
  static const int dHin[5]  = {8, 16, 32, 64, 128};

  char* base = (char*)d_ws;
  size_t off = 0;
  auto carve = [&](size_t bytes) -> void* {
    off = (off + 255) & ~(size_t)255;
    void* p = base + off;
    off += bytes;
    return p;
  };

  u16* wenc[5]; int eKpad[5];
  for (int i = 0; i < 5; ++i) {
    eKpad[i] = eCinP[i] * 16;
    wenc[i] = (u16*)carve(2ull * eCout[i] * eKpad[i]);
  }
  u16* wdec[5]; int dMpad[5];
  for (int i = 0; i < 5; ++i) {
    dMpad[i] = ((dCout[i] + 15) / 16) * 16;
    wdec[i] = (u16*)carve(2ull * 4 * dMpad[i] * (dCin[i] * 4));
  }
  u16* wpre  = (u16*)carve(2ull * 128 * 1024);
  u16* wpost = (u16*)carve(2ull * 1024 * 128);
  u16* wcb   = (u16*)carve(2ull * 512 * 128);

  const long long ACT_ELEMS = 34611200LL;   // 64ch x 32 x 130 x 130
  u16*   actA   = (u16*)carve(2ull * ACT_ELEMS);
  u16*   actB   = (u16*)carve(2ull * ACT_ELEMS);
  float* F      = (float*)carve(4ull * 33554432LL);
  float* qf     = (float*)carve(4ull * 128 * 2048);
  float* scores = (float*)carve(4ull * 512 * 2048);
  float* ccb    = (float*)carve(4ull * 512);
  int*   idx    = (int*)carve(4ull * 2048);
  float* stats  = (float*)carve(4ull * 2 * 1024);
  float* acc2   = (float*)carve(4ull * 2);

  // --- pack all weights to bf16 ---
  for (int i = 0; i < 5; ++i) {
    long long tot = (long long)eCout[i] * eKpad[i];
    int KtotReal = (i == 0 ? 3 : eCinP[i]) * 16;
    pack_conv_k<<<(unsigned)((tot + 255) / 256), 256, 0, stream>>>(
        enc_w[i], wenc[i], eCout[i], KtotReal, eKpad[i], eCout[i]);
  }
  for (int i = 0; i < 5; ++i) {
    long long tot = 4LL * dMpad[i] * (dCin[i] * 4);
    pack_deconv_k<<<(unsigned)((tot + 255) / 256), 256, 0, stream>>>(
        dec_w[i], wdec[i], dCout[i], dMpad[i], dCin[i]);
  }
  pack_conv_k<<<(128 * 1024 + 255) / 256, 256, 0, stream>>>(pre_w,  wpre,  128,  1024, 1024, 128);
  pack_conv_k<<<(1024 * 128 + 255) / 256, 256, 0, stream>>>(post_w, wpost, 1024, 128,  128,  1024);
  pack_conv_k<<<(512 * 128 + 255) / 256, 256, 0, stream>>>(cb_f,   wcb,   512,  128,  128,  512);

  // --- input: zero 4ch halo buffer, scatter x interior (ch 0..2) ---
  {
    long long tot = 4LL * Bn * 258 * 258;
    zero_u16_k<<<(unsigned)((tot + 255) / 256), 256, 0, stream>>>(actA, tot);
    long long n = (long long)Bn * 3 * 65536;
    nchw2halo_k<<<(unsigned)((n + 255) / 256), 256, 0, stream>>>(x, actA, Bn, 3, 256, 256);
  }

  // --- encoder ---
  const u16* cur = actA;
  u16* nxt = actB;
  for (int i = 0; i < 5; ++i) {
    int Hin = eHin[i], Hout = Hin / 2;
    int Nout = Bn * Hout * Hout;
    conv_gemm_k<1><<<dim3((unsigned)(Nout / 64), eCout[i] / 16), 32, 0, stream>>>(
        wenc[i], cur, enc_b[i], F,
        eCout[i], Nout, eKpad[i],
        Bn, Hin, Hin, Hout, Hout, 2, 1);
    bn_stats_k<<<eCout[i], 256, 0, stream>>>(F, stats, Nout);
    if (i < 4) {
      long long halos = (long long)eCout[i] * Bn * (2 * (Hout + 2) + 2 * Hout);
      halo_zero_k<<<(unsigned)((halos + 255) / 256), 256, 0, stream>>>(nxt, eCout[i], Bn, Hout, Hout);
      bn_relu_k<1><<<dim3((unsigned)(Nout / 256), eCout[i]), 256, 0, stream>>>(
          F, stats, enc_g[i], enc_be[i], nxt, Bn, Hout, Hout);
    } else {
      bn_relu_k<0><<<dim3((unsigned)(Nout / 256), eCout[i]), 256, 0, stream>>>(
          F, stats, enc_g[i], enc_be[i], nxt, Bn, Hout, Hout);   // plain [1024][2048]
    }
    const u16* tmp = cur; cur = nxt; nxt = (u16*)tmp;
  }

  // --- pre 1x1 conv -> q f32 [128][2048] ---
  conv_gemm_k<0><<<dim3(2048 / 64, 128 / 16), 32, 0, stream>>>(
      wpre, cur, pre_b, qf, 128, 2048, 1024,
      1, 1, 2048, 1, 2048, 1, 0);
  f32_to_bf16_k<<<(128 * 2048 + 255) / 256, 256, 0, stream>>>(qf, (u16*)actA, 128LL * 2048);

  // --- VQ ---
  cc_k<<<2, 256, 0, stream>>>(cb_f, ccb);
  conv_gemm_k<0><<<dim3(2048 / 64, 512 / 16), 32, 0, stream>>>(
      wcb, (const u16*)actA, (const float*)nullptr, scores, 512, 2048, 128,
      1, 1, 2048, 1, 2048, 1, 0);
  vq_argmin_k<<<8, 256, 0, stream>>>(scores, ccb, idx);
  vq_gather_k<<<(128 * 2048 + 255) / 256, 256, 0, stream>>>(cb_f, idx, actB);
  zero2_k<<<1, 1, 0, stream>>>(acc2);
  vq_loss_k<<<2048, 128, 0, stream>>>(qf, cb_f, idx, acc2);

  // --- post 1x1 conv ---
  conv_gemm_k<0><<<dim3(2048 / 64, 1024 / 16), 32, 0, stream>>>(
      wpost, (const u16*)actB, post_b, F, 1024, 2048, 128,
      1, 1, 2048, 1, 2048, 1, 0);
  {
    long long halos = 1024LL * Bn * (2 * 10 + 2 * 8);
    halo_zero_k<<<(unsigned)((halos + 255) / 256), 256, 0, stream>>>(actA, 1024, Bn, 8, 8);
    f32cn_to_halo_k<<<dim3(2048 / 256, 1024), 256, 0, stream>>>(F, actA, Bn, 8, 8);
  }

  // --- decoder ---
  cur = actA; nxt = actB;
  for (int i = 0; i < 5; ++i) {
    int Hin = dHin[i], Hout = 2 * Hin;
    int Nclass = Bn * Hin * Hin;
    int NoutT  = Bn * Hout * Hout;
    deconv_gemm_k<<<dim3((unsigned)(Nclass / 64), dMpad[i] / 16, 4), 32, 0, stream>>>(
        wdec[i], cur, dec_b[i], F,
        dCout[i], dMpad[i], dCin[i], Bn, Hin, Hin, Hout, Hout);
    bn_stats_k<<<dCout[i], 256, 0, stream>>>(F, stats, NoutT);
    if (i < 4) {
      long long halos = (long long)dCout[i] * Bn * (2 * (Hout + 2) + 2 * Hout);
      halo_zero_k<<<(unsigned)((halos + 255) / 256), 256, 0, stream>>>(nxt, dCout[i], Bn, Hout, Hout);
      bn_relu_k<1><<<dim3((unsigned)(NoutT / 256), dCout[i]), 256, 0, stream>>>(
          F, stats, dec_g[i], dec_be[i], nxt, Bn, Hout, Hout);
      const u16* tmp = cur; cur = nxt; nxt = (u16*)tmp;
    } else {
      bn_sig_out_k<<<dim3((unsigned)(NoutT / 256), 3), 256, 0, stream>>>(
          F, stats, dec_g[i], dec_be[i], (float*)d_out, NoutT, 65536, 3);
    }
  }

  // --- loss scalar ---
  final_loss_k<<<1, 1, 0, stream>>>(acc2, (float*)d_out + 6291456LL);
}
